// AttentionOp_62448824484475
// MI455X (gfx1250) — matrix operand order repaired and checked by reading, not compiled, not run on hardware
//
#include <hip/hip_runtime.h>

// ---------------------------------------------------------------------------
// Flash-attention (causal + segment-id mask, GQA) for MI455X / gfx1250.
// bf16 WMMA matmuls (v_wmma_f32_16x16x32_bf16), f32 online softmax.
// B=2, T=1024, NQ=32, NKV=8 (G=4), D=128.
// ---------------------------------------------------------------------------

typedef __bf16 v16bf __attribute__((ext_vector_type(16)));
typedef float  v8f   __attribute__((ext_vector_type(8)));
typedef float  f32x4 __attribute__((ext_vector_type(4)));

#define AB   2
#define AT   1024
#define ANQ  32
#define ANKV 8
#define AD   128
#define AG   4                      // NQ / NKV
#define MASK_VALUE (-2.3819763e38f) // -0.7 * FLT_MAX (reference DEFAULT_MASK_VALUE)

__global__ __launch_bounds__(256) void fa_gqa_kernel(
    const float* __restrict__ Q, const float* __restrict__ K,
    const float* __restrict__ V, const int* __restrict__ SEG,
    float* __restrict__ O)
{
    // LDS staging: K block (32 keys x 128d) bf16, V block transposed, P scratch.
    __shared__ __bf16 lds_k[32][AD];        // 8 KB
    __shared__ __bf16 lds_vt[AD][32];       // 8 KB (V transposed: [d][key])
    __shared__ __bf16 lds_p[8][16][32];     // 8 KB (per-wave P tile scratch)

    const int tid  = threadIdx.x;
    const int wave = tid >> 5;
    const int lane = tid & 31;
    const int ln   = lane & 15;             // column / row-within-16 selector
    const int hi   = lane >> 4;             // which 16-lane half

    const int qt  = blockIdx.x & 31;            // 32-row q tile index (T/32)
    const int kvh = (blockIdx.x >> 5) & 7;      // kv head
    const int b   = blockIdx.x >> 8;            // batch

    const int g     = wave & 3;                 // head-in-group
    const int h     = kvh * AG + g;             // q head
    const int qbase = qt * 32 + (wave >> 2) * 16;

    // ---- Q tile -> WMMA A layout (16x32 bf16 per K-chunk, 4 chunks) --------
    // A layout: lanes 0-15 & 16-31 both hold rows M=0..15; element j of the
    // v16bf holds K = hi*16 + j  (+ chunk*32).
    v16bf qa[4];
    {
        const long qrow = ((long)(b * AT + qbase + ln) * ANQ + h) * AD;
#pragma unroll
        for (int c = 0; c < 4; ++c) {
            const float* p0 = Q + qrow + c * 32 + hi * 16;      // K = c*32+hi*16+0..7
            const float* p1 = p0 + 8;                           // K = +8
            f32x4 a0 = *(const f32x4*)(p0);
            f32x4 a1 = *(const f32x4*)(p0 + 4);
            f32x4 b0 = *(const f32x4*)(p1);
            f32x4 b1 = *(const f32x4*)(p1 + 4);
#pragma unroll
            for (int j = 0; j < 4; ++j) {
                qa[c][j]      = (__bf16)a0[j];
                qa[c][4 + j]  = (__bf16)a1[j];
                qa[c][8 + j]  = (__bf16)b0[j];
                qa[c][12 + j] = (__bf16)b1[j];
            }
        }
    }

    // Per-row (C-layout row = r + 8*hi) metadata.
    int trow[8], rseg[8];
#pragma unroll
    for (int r = 0; r < 8; ++r) {
        trow[r] = qbase + r + 8 * hi;
        rseg[r] = SEG[b * AT + trow[r]];
    }

    // Online-softmax state.
    v8f m_i, l_i, o_acc[8];
#pragma unroll
    for (int r = 0; r < 8; ++r) { m_i[r] = -__builtin_inff(); l_i[r] = 0.0f; }
#pragma unroll
    for (int dt = 0; dt < 8; ++dt) o_acc[dt] = (v8f)(0.0f);

    const int nkb = qt + 1;  // uniform across the workgroup -> barriers are safe
    for (int kb = 0; kb < nkb; ++kb) {
        // ---- cooperative stage of K/V block (32 keys x 128) as bf16 --------
        {
            const int key = tid >> 3;           // 0..31
            const int d0  = (tid & 7) * 16;     // 0..112
            const long src = ((long)(b * AT + kb * 32 + key) * ANKV + kvh) * AD + d0;
            const float* kp = K + src;
            const float* vp = V + src;
#pragma unroll
            for (int j = 0; j < 16; j += 4) {
                f32x4 kk = *(const f32x4*)(kp + j);
                f32x4 vv = *(const f32x4*)(vp + j);
#pragma unroll
                for (int e = 0; e < 4; ++e) {
                    lds_k[key][d0 + j + e]  = (__bf16)kk[e];
                    lds_vt[d0 + j + e][key] = (__bf16)vv[e];
                }
            }
            if (kb + 1 < nkb) {                 // prefetch next key block
                __builtin_prefetch(kp + (long)32 * ANKV * AD, 0, 1);
                __builtin_prefetch(vp + (long)32 * ANKV * AD, 0, 1);
            }
        }
        __syncthreads();

        // ---- S = Q K^T : two 16x16 tiles, each 4 chained K32 WMMAs ---------
        v8f st[2];
#pragma unroll
        for (int nt = 0; nt < 2; ++nt) {
            v8f acc = (v8f)(0.0f);
            const int key_local = nt * 16 + ln;     // B-layout column = key
#pragma unroll
            for (int c = 0; c < 4; ++c) {
                // B layout: element j holds K-dim value d = c*32 + hi*16 + j.
                v16bf bb = *(const v16bf*)&lds_k[key_local][c * 32 + hi * 16];
                acc = __builtin_amdgcn_wmma_f32_16x16x32_bf16(
                    false, qa[c], false, bb, (short)0, acc, false, false);
            }
            // causal + segment mask (reference: where(mask, logits, MASK)).
            const int s_col = kb * 32 + nt * 16 + ln;
            const int cseg  = SEG[b * AT + s_col];
#pragma unroll
            for (int r = 0; r < 8; ++r) {
                const bool act = (s_col <= trow[r]) && (cseg == rseg[r]);
                acc[r] = act ? acc[r] : MASK_VALUE;
            }
            st[nt] = acc;
        }

        // ---- online softmax update (row = 16 lanes of one half) ------------
        v8f m_new, alpha;
#pragma unroll
        for (int r = 0; r < 8; ++r) {
            float vmax = fmaxf(st[0][r], st[1][r]);
            vmax = fmaxf(vmax, __shfl_xor(vmax, 1, 16));
            vmax = fmaxf(vmax, __shfl_xor(vmax, 2, 16));
            vmax = fmaxf(vmax, __shfl_xor(vmax, 4, 16));
            vmax = fmaxf(vmax, __shfl_xor(vmax, 8, 16));
            m_new[r] = fmaxf(m_i[r], vmax);
            alpha[r] = __expf(m_i[r] - m_new[r]);   // exp(-inf)=0 on first block
            m_i[r]   = m_new[r];
        }
#pragma unroll
        for (int nt = 0; nt < 2; ++nt)
#pragma unroll
            for (int r = 0; r < 8; ++r)
                st[nt][r] = __expf(st[nt][r] - m_new[r]);  // masked -> 0
#pragma unroll
        for (int r = 0; r < 8; ++r) {
            float s = st[0][r] + st[1][r];
            s += __shfl_xor(s, 1, 16);
            s += __shfl_xor(s, 2, 16);
            s += __shfl_xor(s, 4, 16);
            s += __shfl_xor(s, 8, 16);
            l_i[r] = l_i[r] * alpha[r] + s;
        }
#pragma unroll
        for (int dt = 0; dt < 8; ++dt)
#pragma unroll
            for (int r = 0; r < 8; ++r)
                o_acc[dt][r] *= alpha[r];

        // ---- P: C layout -> bf16 A layout via wave-private LDS bounce ------
#pragma unroll
        for (int nt = 0; nt < 2; ++nt)
#pragma unroll
            for (int r = 0; r < 8; ++r)
                lds_p[wave][r + 8 * hi][nt * 16 + ln] = (__bf16)st[nt][r];
        // wave-internal cross-lane LDS RAW: drain the DS counter.
        asm volatile("s_wait_dscnt 0" ::: "memory");
        v16bf pa;
        {
            const __bf16* pr = &lds_p[wave][ln][0];
#pragma unroll
            for (int j = 0; j < 8; ++j) {
                pa[j]     = pr[hi * 16 + j];       // keys hi*16 + 0..7
                pa[8 + j] = pr[hi * 16 + 8 + j];   // keys hi*16 + 8..15
            }
        }

        // ---- O += P V : 8 N-tiles of 16 over D=128 -------------------------
#pragma unroll
        for (int dt = 0; dt < 8; ++dt) {
            // B layout: column n = d-within-tile, element j -> key hi*16 + j.
            v16bf vb = *(const v16bf*)&lds_vt[dt * 16 + ln][hi * 16];
            o_acc[dt] = __builtin_amdgcn_wmma_f32_16x16x32_bf16(
                false, pa, false, vb, (short)0, o_acc[dt], false, false);
        }
        __syncthreads();   // protect LDS K/V before next block's staging
    }

    // ---- epilogue: normalize and store (f32 output) ------------------------
#pragma unroll
    for (int r = 0; r < 8; ++r) {
        const float inv = 1.0f / l_i[r];           // diag always active -> l>0
        const long obase = ((long)(b * AT + trow[r]) * ANQ + h) * AD;
#pragma unroll
        for (int dt = 0; dt < 8; ++dt)
            O[obase + dt * 16 + ln] = o_acc[dt][r] * inv;
    }
}

extern "C" void kernel_launch(void* const* d_in, const int* in_sizes, int n_in,
                              void* d_out, int out_size, void* d_ws, size_t ws_size,
                              hipStream_t stream) {
    (void)in_sizes; (void)n_in; (void)out_size; (void)d_ws; (void)ws_size;
    const float* Q   = (const float*)d_in[0];
    const float* K   = (const float*)d_in[1];
    const float* V   = (const float*)d_in[2];
    const int*   SEG = (const int*)d_in[3];
    float* O = (float*)d_out;
    // grid: B * NKV * (T/32) = 2 * 8 * 32 = 512 blocks of 256 threads (8 waves)
    dim3 grid(AB * ANKV * (AT / 32));
    fa_gqa_kernel<<<grid, 256, 0, stream>>>(Q, K, V, SEG, O);
}
